// STTLayer_61770219651231
// MI455X (gfx1250) — matrix operand order repaired and checked by reading, not compiled, not run on hardware
//
#include <hip/hip_runtime.h>
#include <hip/hip_bf16.h>
#include <stdint.h>

// ---------------- problem dims ----------------
constexpr int NB  = 4;
constexpr int NT  = 4096;
constexpr int ND  = 2048;
constexpr int NH  = 16;
constexpr int NHD = 128;
constexpr int NFF = 8192;
constexpr int NK  = 1024;          // CAP * T
constexpr float EPSV  = 1e-6f;
constexpr float THETA = 10000.0f;

typedef __attribute__((ext_vector_type(16))) __bf16 v16bf;
typedef __attribute__((ext_vector_type(8)))  float  v8f;
typedef __attribute__((ext_vector_type(4)))  int    v4i;

static __device__ inline v8f zero8() {
  v8f z = {0.f,0.f,0.f,0.f,0.f,0.f,0.f,0.f};
  return z;
}

static __device__ inline v8f wmma_bf16(v16bf a, v16bf b, v8f c) {
  return __builtin_amdgcn_wmma_f32_16x16x32_bf16(false, a, false, b, (short)0, c, false, false);
}

union FragU { uint4 u[2]; v16bf v; };

// ---------------- CDNA5 async global->LDS path (guarded, with sync fallback) ----
#if defined(__has_builtin)
#if __has_builtin(__builtin_amdgcn_global_load_async_to_lds_b128) && \
    __has_builtin(__builtin_amdgcn_s_wait_asynccnt)
#define USE_ASYNC_LDS 1
#endif
#endif
#ifndef USE_ASYNC_LDS
#define USE_ASYNC_LDS 0
#endif

typedef __attribute__((address_space(1))) v4i as1_v4i;
typedef __attribute__((address_space(3))) v4i as3_v4i;

// copy 16 bytes global -> LDS
static __device__ inline void ld_lds_b128(const __bf16* g, __bf16* lds) {
#if USE_ASYNC_LDS
  __builtin_amdgcn_global_load_async_to_lds_b128(
      (as1_v4i*)(uintptr_t)g,
      (as3_v4i*)(uint32_t)(uintptr_t)lds,   // generic LDS ptr low 32 bits = LDS offset
      0, 0);
#else
  *(uint4*)lds = *(const uint4*)g;
#endif
}

static __device__ inline void lds_copy_wait() {
#if USE_ASYNC_LDS
  __builtin_amdgcn_s_wait_asynccnt(0);
#endif
}

// ---------------- 1) f32 -> bf16 convert ----------------
__global__ void cvt_bf16_kernel(const float* __restrict__ src,
                                __bf16* __restrict__ dst, size_t n) {
  size_t i = (size_t)blockIdx.x * blockDim.x + threadIdx.x;
  if (i < n) dst[i] = (__bf16)src[i];
}

// ---------------- 2) router logits ----------------
__global__ void router_kernel(const float* __restrict__ hs,
                              const float* __restrict__ wr,
                              const float* __restrict__ br,
                              float* __restrict__ logits) {
  int wid  = threadIdx.x >> 5;
  int lane = threadIdx.x & 31;
  int tok  = blockIdx.x * 8 + wid;               // 8 waves per 256-thread block
  const float* row = hs + (size_t)tok * ND;
  float s = 0.f;
  for (int i = lane; i < ND; i += 32) s += row[i] * wr[i];
  #pragma unroll
  for (int m = 16; m >= 1; m >>= 1) s += __shfl_xor(s, m, 32);
  if (lane == 0) logits[tok] = s + br[0];
}

// ---------------- 3) top-K select + ascending sort ----------------
__global__ __launch_bounds__(1024) void topk_kernel(const float* __restrict__ logits,
                                                    int* __restrict__ pos) {
  __shared__ float key[NT];
  __shared__ int   idx[NT];
  __shared__ int   sel[NK];
  int b = blockIdx.x;
  for (int i = threadIdx.x; i < NT; i += blockDim.x) {
    key[i] = logits[b * NT + i];
    idx[i] = i;
  }
  __syncthreads();
  // bitonic sort, descending by key (tie: smaller index first)
  for (int k = 2; k <= NT; k <<= 1) {
    for (int j = k >> 1; j > 0; j >>= 1) {
      for (int i = threadIdx.x; i < NT; i += blockDim.x) {
        int ixj = i ^ j;
        if (ixj > i) {
          bool desc = ((i & k) == 0);
          float ka = key[i], kb2 = key[ixj];
          int ia = idx[i], ib = idx[ixj];
          bool sw = desc ? (ka < kb2 || (ka == kb2 && ia > ib))
                         : (ka > kb2 || (ka == kb2 && ia < ib));
          if (sw) { key[i] = kb2; key[ixj] = ka; idx[i] = ib; idx[ixj] = ia; }
        }
      }
      __syncthreads();
    }
  }
  for (int i = threadIdx.x; i < NK; i += blockDim.x) sel[i] = idx[i];
  __syncthreads();
  // bitonic sort ascending on the K winning indices
  for (int k = 2; k <= NK; k <<= 1) {
    for (int j = k >> 1; j > 0; j >>= 1) {
      for (int i = threadIdx.x; i < NK; i += blockDim.x) {
        int ixj = i ^ j;
        if (ixj > i) {
          bool asc = ((i & k) == 0);
          int a = sel[i], c = sel[ixj];
          bool sw = asc ? (a > c) : (a < c);
          if (sw) { sel[i] = c; sel[ixj] = a; }
        }
      }
      __syncthreads();
    }
  }
  for (int i = threadIdx.x; i < NK; i += blockDim.x) pos[b * NK + i] = sel[i];
}

// ---------------- 4) (gather +) RMSNorm -> bf16 ----------------
__global__ void rms_kernel(const float* __restrict__ src,
                           const int* __restrict__ pos,   // null => direct rows
                           const float* __restrict__ w,
                           float* __restrict__ xout,      // optional raw copy (f32)
                           __bf16* __restrict__ hout) {
  int r = blockIdx.x;                       // row in [0, NB*NK)
  const float* row;
  if (pos) {
    int b = r / NK;
    row = src + ((size_t)b * NT + pos[r]) * ND;
  } else {
    row = src + (size_t)r * ND;
  }
  __shared__ float red[256];
  float ss = 0.f;
  for (int i = threadIdx.x; i < ND; i += 256) { float v = row[i]; ss += v * v; }
  red[threadIdx.x] = ss;
  __syncthreads();
  for (int s = 128; s > 0; s >>= 1) {
    if (threadIdx.x < s) red[threadIdx.x] += red[threadIdx.x + s];
    __syncthreads();
  }
  float rs = rsqrtf(red[0] / (float)ND + EPSV);
  for (int i = threadIdx.x; i < ND; i += 256) {
    float v = row[i];
    if (xout) xout[(size_t)r * ND + i] = v;
    hout[(size_t)r * ND + i] = (__bf16)(v * rs * w[i]);
  }
}

// ---------------- 5) tiled bf16 WMMA GEMM with fused epilogues ----------------
// C(M,N) = A(M,Kd) @ W(Kd,N), block tile 128x256x32, 8 waves, 64x64 per wave.
// MODE 0: outf = acc + bias[n]
// MODE 1: outf = res  + acc
// MODE 4: outf[(b*NT + pos[m])*ND + n] = res + acc   (scatter epilogue)
template <int MODE>
__global__ __launch_bounds__(256) void gemm_kernel(
    const __bf16* __restrict__ A, const __bf16* __restrict__ W,
    const float* __restrict__ bias, const float* __restrict__ res,
    const int* __restrict__ pos,
    float* __restrict__ outf,
    int M, int N, int Kd) {
  __shared__ __align__(16) __bf16 As[128 * 32];
  __shared__ __align__(16) __bf16 Bs[32 * 256];

  int m0 = blockIdx.x * 128;
  int n0 = blockIdx.y * 256;
  int tid = threadIdx.x;
  int wid = tid >> 5, lane = tid & 31;
  int wm = (wid >> 2) * 64;                 // 0 or 64
  int wn = (wid & 3) * 64;                  // 0,64,128,192
  int half = lane >> 4, lm = lane & 15;

  v8f acc[4][4];
  #pragma unroll
  for (int i = 0; i < 4; i++)
    #pragma unroll
    for (int j = 0; j < 4; j++) acc[i][j] = zero8();

  int rowa = tid >> 1;                      // 0..127
  int cola = (tid & 1) * 16;
  int rowb = tid >> 3;                      // 0..31
  int colb = (tid & 7) * 32;

  for (int k0 = 0; k0 < Kd; k0 += 32) {
    {
      const __bf16* g = A + (size_t)(m0 + rowa) * Kd + k0 + cola;
      __bf16* s = As + rowa * 32 + cola;
      ld_lds_b128(g, s);
      ld_lds_b128(g + 8, s + 8);
      if (k0 + 32 < Kd) __builtin_prefetch(g + 32, 0, 3);
    }
    {
      const __bf16* g = W + (size_t)(k0 + rowb) * N + n0 + colb;
      __bf16* s = Bs + rowb * 256 + colb;
      ld_lds_b128(g, s);
      ld_lds_b128(g + 8,  s + 8);
      ld_lds_b128(g + 16, s + 16);
      ld_lds_b128(g + 24, s + 24);
      if (k0 + 32 < Kd) __builtin_prefetch(g + (size_t)32 * N, 0, 3);
    }
    lds_copy_wait();
    __syncthreads();

    v16bf af[4], bfr[4];
    #pragma unroll
    for (int i = 0; i < 4; i++) {
      int mr = wm + i * 16 + lm;
      FragU u;
      u.u[0] = *(const uint4*)(As + mr * 32 + half * 8);
      u.u[1] = *(const uint4*)(As + mr * 32 + 16 + half * 8);
      af[i] = u.v;
    }
    #pragma unroll
    for (int j = 0; j < 4; j++) {
      v16bf t;
      const __bf16* p = Bs + wn + j * 16 + lm;
      #pragma unroll
      for (int s = 0; s < 16; s++)
        t[s] = p[(half * 8 + (s & 7) + (s >> 3) * 16) * 256];
      bfr[j] = t;
    }
    #pragma unroll
    for (int i = 0; i < 4; i++)
      #pragma unroll
      for (int j = 0; j < 4; j++)
        acc[i][j] = wmma_bf16(af[i], bfr[j], acc[i][j]);
    __syncthreads();
  }

  // epilogue: element (i,j,r): m = half*8+r (C layout), n = lm
  #pragma unroll
  for (int i = 0; i < 4; i++) {
    #pragma unroll
    for (int j = 0; j < 4; j++) {
      #pragma unroll
      for (int r = 0; r < 8; r++) {
        int m = m0 + wm + i * 16 + half * 8 + r;
        int n = n0 + wn + j * 16 + lm;
        float v = acc[i][j][r];
        size_t off = (size_t)m * N + n;
        if (MODE == 0) outf[off] = v + bias[n];
        if (MODE == 1) outf[off] = res[off] + v;
        if (MODE == 4) {
          int b = m / NK;
          int t = pos[m];
          outf[((size_t)b * NT + t) * ND + n] = res[(size_t)m * ND + n] + v;
        }
      }
    }
  }
}

// ---------------- 5b) fused gate/up GEMM: act = silu(A@Wg) * (A@Wu) -> bf16 -----
// block tile 128x128x32, 8 waves (2x4), wave tile 64x32, dual accumulators.
__global__ __launch_bounds__(256) void gateup_kernel(
    const __bf16* __restrict__ A, const __bf16* __restrict__ Wg,
    const __bf16* __restrict__ Wu, __bf16* __restrict__ act,
    int M, int N, int Kd) {
  __shared__ __align__(16) __bf16 As[128 * 32];
  __shared__ __align__(16) __bf16 Gs[32 * 128];
  __shared__ __align__(16) __bf16 Us[32 * 128];

  int m0 = blockIdx.x * 128;
  int n0 = blockIdx.y * 128;
  int tid = threadIdx.x;
  int wid = tid >> 5, lane = tid & 31;
  int wm = (wid >> 2) * 64;                 // 0 or 64
  int wn = (wid & 3) * 32;                  // 0,32,64,96
  int half = lane >> 4, lm = lane & 15;

  v8f ag[4][2], au[4][2];
  #pragma unroll
  for (int i = 0; i < 4; i++)
    #pragma unroll
    for (int j = 0; j < 2; j++) { ag[i][j] = zero8(); au[i][j] = zero8(); }

  int rowa = tid >> 1;                      // 0..127
  int cola = (tid & 1) * 16;
  int rowb = tid >> 3;                      // 0..31
  int colb = (tid & 7) * 16;

  for (int k0 = 0; k0 < Kd; k0 += 32) {
    {
      const __bf16* g = A + (size_t)(m0 + rowa) * Kd + k0 + cola;
      __bf16* s = As + rowa * 32 + cola;
      ld_lds_b128(g, s);
      ld_lds_b128(g + 8, s + 8);
    }
    {
      size_t goff = (size_t)(k0 + rowb) * N + n0 + colb;
      const __bf16* gg = Wg + goff;
      const __bf16* gu = Wu + goff;
      __bf16* sg = Gs + rowb * 128 + colb;
      __bf16* su = Us + rowb * 128 + colb;
      ld_lds_b128(gg, sg);
      ld_lds_b128(gg + 8, sg + 8);
      ld_lds_b128(gu, su);
      ld_lds_b128(gu + 8, su + 8);
    }
    lds_copy_wait();
    __syncthreads();

    v16bf af[4], bg[2], bu[2];
    #pragma unroll
    for (int i = 0; i < 4; i++) {
      int mr = wm + i * 16 + lm;
      FragU u;
      u.u[0] = *(const uint4*)(As + mr * 32 + half * 8);
      u.u[1] = *(const uint4*)(As + mr * 32 + 16 + half * 8);
      af[i] = u.v;
    }
    #pragma unroll
    for (int j = 0; j < 2; j++) {
      v16bf t1, t2;
      const __bf16* pg = Gs + wn + j * 16 + lm;
      const __bf16* pu = Us + wn + j * 16 + lm;
      #pragma unroll
      for (int s = 0; s < 16; s++) {
        int kk = half * 8 + (s & 7) + (s >> 3) * 16;
        t1[s] = pg[kk * 128];
        t2[s] = pu[kk * 128];
      }
      bg[j] = t1;
      bu[j] = t2;
    }
    #pragma unroll
    for (int i = 0; i < 4; i++)
      #pragma unroll
      for (int j = 0; j < 2; j++) {
        ag[i][j] = wmma_bf16(af[i], bg[j], ag[i][j]);
        au[i][j] = wmma_bf16(af[i], bu[j], au[i][j]);
      }
    __syncthreads();
  }

  #pragma unroll
  for (int i = 0; i < 4; i++) {
    #pragma unroll
    for (int j = 0; j < 2; j++) {
      #pragma unroll
      for (int r = 0; r < 8; r++) {
        int m = m0 + wm + i * 16 + half * 8 + r;
        int n = n0 + wn + j * 16 + lm;
        float g = ag[i][j][r];
        float u = au[i][j][r];
        float sg = g / (1.f + __expf(-g));
        act[(size_t)m * N + n] = (__bf16)(sg * u);
      }
    }
  }
}

// ---------------- 6) RoPE: q -> bf16 row-major, k -> bf16 transposed (B,H,HD,K) ----
__global__ void rope_kernel(const float* __restrict__ qf, const float* __restrict__ kf,
                            const int* __restrict__ pos,
                            __bf16* __restrict__ qb, __bf16* __restrict__ kT) {
  int gid = blockIdx.x;                 // r*NH + h
  int h = gid % NH;
  int r = gid / NH;                     // row in [0, NB*NK)
  int b = r / NK;
  int kidx = r % NK;
  int i = threadIdx.x;                  // 0..127
  float p = (float)pos[r];
  int d = i & 63;
  float inv = __powf(THETA, -((float)(2 * d) / (float)NHD));
  float ang = p * inv;
  float c = __cosf(ang), s = __sinf(ang);
  size_t base = (size_t)r * ND + h * NHD;
  float q = qf[base + i];
  float k = kf[base + i];
  float qrot = (i < 64) ? -qf[base + i + 64] : qf[base + i - 64];
  float krot = (i < 64) ? -kf[base + i + 64] : kf[base + i - 64];
  qb[base + i] = (__bf16)(q * c + qrot * s);
  kT[((size_t)(b * NH + h) * NHD + i) * NK + kidx] = (__bf16)(k * c + krot * s);
}

// ---------------- 7) flash attention (online softmax), bf16 WMMA ----------------
// grid = (NB*NH, NK/128), block = 256 (8 waves). Wave handles 16 q rows.
__global__ __launch_bounds__(256) void attn_kernel(
    const __bf16* __restrict__ qb, const __bf16* __restrict__ kT,
    const __bf16* __restrict__ vb, const int* __restrict__ pos,
    __bf16* __restrict__ ctxb) {
  int bh = blockIdx.x;
  int b = bh / NH, h = bh % NH;
  int wid = threadIdx.x >> 5, lane = threadIdx.x & 31;
  int q0 = blockIdx.y * 128 + wid * 16;
  int half = lane >> 4, lm = lane & 15;
  const float scale = 0.08838834764831845f;   // 1/sqrt(128)

  __shared__ __align__(16) __bf16 plds[8][16 * 32];

  // preload Q fragments (A-matrix: row = lm)
  v16bf aq[4];
  {
    const __bf16* qrow = qb + ((size_t)(b * NK + q0 + lm) * ND + h * NHD);
    #pragma unroll
    for (int c = 0; c < 4; c++) {
      FragU u;
      u.u[0] = *(const uint4*)(qrow + c * 32 + half * 8);
      u.u[1] = *(const uint4*)(qrow + c * 32 + 16 + half * 8);
      aq[c] = u.v;
    }
  }
  int qpos[8];
  #pragma unroll
  for (int r = 0; r < 8; r++) qpos[r] = pos[b * NK + q0 + half * 8 + r];

  float mrow[8], lrow[8];
  #pragma unroll
  for (int r = 0; r < 8; r++) { mrow[r] = -1e30f; lrow[r] = 0.f; }
  v8f oacc[8];
  #pragma unroll
  for (int t = 0; t < 8; t++) oacc[t] = zero8();

  const __bf16* kbase = kT + (size_t)bh * NHD * NK;

  for (int j = 0; j < NK / 32; j++) {
    v8f s0 = zero8(), s1 = zero8();
    #pragma unroll
    for (int c = 0; c < 4; c++) {
      // B-fragment: lane = token column (lm), elements = dims in table k order
      const __bf16* kp = kbase + (size_t)(c * 32) * NK + j * 32 + lm;
      v16bf u0, u1;
      #pragma unroll
      for (int s = 0; s < 16; s++) {
        int kk = half * 8 + (s & 7) + (s >> 3) * 16;
        u0[s] = kp[(size_t)kk * NK];
        u1[s] = kp[(size_t)kk * NK + 16];
      }
      s0 = wmma_bf16(aq[c], u0, s0);
      s1 = wmma_bf16(aq[c], u1, s1);
    }
    int kp0 = pos[b * NK + j * 32 + lm];
    int kp1 = pos[b * NK + j * 32 + 16 + lm];
    #pragma unroll
    for (int r = 0; r < 8; r++) {
      float a0 = (qpos[r] >= kp0) ? s0[r] * scale : -1e30f;
      float a1 = (qpos[r] >= kp1) ? s1[r] * scale : -1e30f;
      float rmax = fmaxf(a0, a1);
      #pragma unroll
      for (int msk = 1; msk < 16; msk <<= 1)
        rmax = fmaxf(rmax, __shfl_xor(rmax, msk, 32));
      float mnew = fmaxf(mrow[r], rmax);
      float alpha = (mnew <= -1e29f) ? 1.f : __expf(mrow[r] - mnew);
      float p0 = (a0 <= -1e29f) ? 0.f : __expf(a0 - mnew);
      float p1 = (a1 <= -1e29f) ? 0.f : __expf(a1 - mnew);
      float ps = p0 + p1;
      #pragma unroll
      for (int msk = 1; msk < 16; msk <<= 1)
        ps += __shfl_xor(ps, msk, 32);
      lrow[r] = lrow[r] * alpha + ps;
      mrow[r] = mnew;
      #pragma unroll
      for (int t = 0; t < 8; t++) oacc[t][r] *= alpha;
      int m = half * 8 + r;
      plds[wid][m * 32 + lm]      = (__bf16)p0;
      plds[wid][m * 32 + 16 + lm] = (__bf16)p1;
    }
    __syncthreads();
    // ctx += P(16x32) @ V(32x128)
    v16bf ap;
    {
      FragU u;
      const __bf16* pp = &plds[wid][lm * 32 + half * 8];
      u.u[0] = *(const uint4*)(pp);
      u.u[1] = *(const uint4*)(pp + 16);
      ap = u.v;
    }
    const __bf16* vrow = vb + ((size_t)(b * NK + j * 32) * ND + h * NHD + lm);
    #pragma unroll
    for (int t = 0; t < 8; t++) {
      v16bf u;
      #pragma unroll
      for (int s = 0; s < 16; s++)
        u[s] = vrow[(size_t)(half * 8 + (s & 7) + (s >> 3) * 16) * ND + t * 16];
      oacc[t] = wmma_bf16(ap, u, oacc[t]);
    }
    __syncthreads();
  }

  #pragma unroll
  for (int t = 0; t < 8; t++) {
    #pragma unroll
    for (int r = 0; r < 8; r++) {
      float l = lrow[r];
      float o = (l > 0.f) ? oacc[t][r] / l : 0.f;
      int m = q0 + half * 8 + r;
      ctxb[((size_t)(b * NK + m)) * ND + h * NHD + t * 16 + lm] = (__bf16)o;
    }
  }
}

// ---------------- launcher ----------------
extern "C" void kernel_launch(void* const* d_in, const int* in_sizes, int n_in,
                              void* d_out, int out_size, void* d_ws, size_t ws_size,
                              hipStream_t stream) {
  const float* hidden = (const float*)d_in[0];
  const float* w_router = (const float*)d_in[1];
  const float* b_router = (const float*)d_in[2];
  const float* ln1_w = (const float*)d_in[3];
  const float* wq = (const float*)d_in[4];
  const float* bq = (const float*)d_in[5];
  const float* wk = (const float*)d_in[6];
  const float* bk = (const float*)d_in[7];
  const float* wv = (const float*)d_in[8];
  const float* bv = (const float*)d_in[9];
  const float* wo = (const float*)d_in[10];
  const float* ln2_w = (const float*)d_in[11];
  const float* w_gate = (const float*)d_in[12];
  const float* w_up = (const float*)d_in[13];
  const float* w_down = (const float*)d_in[14];
  float* out = (float*)d_out;

  // workspace carve-out
  uint8_t* wp = (uint8_t*)d_ws;
  auto alloc = [&](size_t bytes) -> void* {
    void* p = (void*)wp;
    wp += (bytes + 255) & ~(size_t)255;
    return p;
  };
  const size_t R = (size_t)NB * NK;          // 4096 selected rows total
  float*  logits = (float*)alloc((size_t)NB * NT * 4);
  int*    pos    = (int*)  alloc((size_t)NB * NK * 4);
  float*  xg     = (float*)alloc(R * ND * 4);
  float*  x2     = (float*)alloc(R * ND * 4);
  float*  qf     = (float*)alloc(R * ND * 4);
  float*  kf     = (float*)alloc(R * ND * 4);
  float*  vf     = (float*)alloc(R * ND * 4);
  __bf16* hb     = (__bf16*)alloc(R * ND * 2);
  __bf16* h2b    = (__bf16*)alloc(R * ND * 2);
  __bf16* qbuf   = (__bf16*)alloc(R * ND * 2);
  __bf16* kTbuf  = (__bf16*)alloc(R * ND * 2);
  __bf16* vbuf   = (__bf16*)alloc(R * ND * 2);
  __bf16* ctxb   = (__bf16*)alloc(R * ND * 2);
  __bf16* actb   = (__bf16*)alloc(R * NFF * 2);
  __bf16* wqb    = (__bf16*)alloc((size_t)ND * ND * 2);
  __bf16* wkb    = (__bf16*)alloc((size_t)ND * ND * 2);
  __bf16* wvb    = (__bf16*)alloc((size_t)ND * ND * 2);
  __bf16* wob    = (__bf16*)alloc((size_t)ND * ND * 2);
  __bf16* wgb    = (__bf16*)alloc((size_t)ND * NFF * 2);
  __bf16* wub    = (__bf16*)alloc((size_t)ND * NFF * 2);
  __bf16* wdb    = (__bf16*)alloc((size_t)NFF * ND * 2);

  // out = hidden (scatter overwrites selected rows later)
  (void)hipMemcpyAsync(out, hidden, (size_t)NB * NT * ND * sizeof(float),
                       hipMemcpyDeviceToDevice, stream);

  // weight conversion f32 -> bf16
  auto cvt = [&](const float* s, __bf16* d, size_t n) {
    cvt_bf16_kernel<<<(unsigned)((n + 255) / 256), 256, 0, stream>>>(s, d, n);
  };
  cvt(wq, wqb, (size_t)ND * ND);
  cvt(wk, wkb, (size_t)ND * ND);
  cvt(wv, wvb, (size_t)ND * ND);
  cvt(wo, wob, (size_t)ND * ND);
  cvt(w_gate, wgb, (size_t)ND * NFF);
  cvt(w_up,   wub, (size_t)ND * NFF);
  cvt(w_down, wdb, (size_t)NFF * ND);

  // router + top-K
  router_kernel<<<NB * NT / 8, 256, 0, stream>>>(hidden, w_router, b_router, logits);
  topk_kernel<<<NB, 1024, 0, stream>>>(logits, pos);

  // gather + RMSNorm -> hb (bf16), keep raw x in xg
  rms_kernel<<<(unsigned)R, 256, 0, stream>>>(hidden, pos, ln1_w, xg, hb);

  // QKV projections (M=4096, N=2048, K=2048)
  dim3 gproj(32, 8);
  gemm_kernel<0><<<gproj, 256, 0, stream>>>(hb, wqb, bq, nullptr, nullptr,
                                            qf, (int)R, ND, ND);
  gemm_kernel<0><<<gproj, 256, 0, stream>>>(hb, wkb, bk, nullptr, nullptr,
                                            kf, (int)R, ND, ND);
  gemm_kernel<0><<<gproj, 256, 0, stream>>>(hb, wvb, bv, nullptr, nullptr,
                                            vf, (int)R, ND, ND);

  // RoPE (q row-major bf16, k transposed bf16) + v -> bf16
  rope_kernel<<<(unsigned)(R * NH), NHD, 0, stream>>>(qf, kf, pos, qbuf, kTbuf);
  cvt(vf, vbuf, R * ND);

  // attention
  attn_kernel<<<dim3(NB * NH, NK / 128), 256, 0, stream>>>(qbuf, kTbuf, vbuf, pos, ctxb);

  // x2 = xg + ctx @ wo
  gemm_kernel<1><<<gproj, 256, 0, stream>>>(ctxb, wob, nullptr, xg, nullptr,
                                            x2, (int)R, ND, ND);

  // RMSNorm 2 -> h2b
  rms_kernel<<<(unsigned)R, 256, 0, stream>>>(x2, nullptr, ln2_w, nullptr, h2b);

  // MLP: fused act = silu(h2@wg) * (h2@wu), then y = x2 + act @ w_down (scattered)
  gateup_kernel<<<dim3(32, NFF / 128), 256, 0, stream>>>(h2b, wgb, wub, actb,
                                                         (int)R, NFF, ND);
  gemm_kernel<4><<<gproj, 256, 0, stream>>>(actb, wdb, nullptr, x2, pos,
                                            out, (int)R, ND, NFF);
}
